// MixedLoraModel_734
// MI455X (gfx1250) — compile-verified, matrix operand order checked by reading, not run on hardware
//
#include <hip/hip_runtime.h>
#include <stdint.h>

#define T_TOK   8192
#define D_IN    4096
#define D_OUT   4096
#define L_ADAPT 8
#define R_RANK  16

#define BM 128          // block tile rows (tokens)
#define BN 256          // block tile cols (d_out)
#define BK 64           // K slab
#define LDSS 72         // LDS row stride in bf16 elems (144B: 16B aligned, conflict-free)
#define XELEMS (BM * LDSS)            // 9216 halfs
#define BUFE   ((BM + BN) * LDSS)     // 27648 halfs per buffer (55296 B)

typedef __attribute__((ext_vector_type(16))) __bf16  v16bf;
typedef __attribute__((ext_vector_type(8)))  float   v8f;
typedef __attribute__((ext_vector_type(4)))  unsigned int   v4u;
typedef __attribute__((ext_vector_type(4)))  float   v4f;
typedef __attribute__((ext_vector_type(4)))  unsigned short v4us;

typedef int b128_t __attribute__((vector_size(16)));   // matches builtin param type

union FragBF { v4u u[2]; v16bf v; };

#ifndef __has_builtin
#define __has_builtin(x) 0
#endif
#if __has_builtin(__builtin_amdgcn_global_load_async_to_lds_b128) && \
    __has_builtin(__builtin_amdgcn_s_wait_asynccnt)
#define USE_ASYNC_LDS 1
#else
#define USE_ASYNC_LDS 0
#endif

__device__ __forceinline__ void cp16_to_lds(const unsigned short* g, unsigned short* l) {
#if USE_ASYNC_LDS
  __builtin_amdgcn_global_load_async_to_lds_b128(
      (__attribute__((address_space(1))) b128_t*)g,
      (__attribute__((address_space(3))) b128_t*)l, 0, 0);
#else
  *(v4u*)l = *(const v4u*)g;
#endif
}
__device__ __forceinline__ void cp_wait_all() {
#if USE_ASYNC_LDS
  __builtin_amdgcn_s_wait_asynccnt(0);
#endif
}

__device__ __forceinline__ unsigned short f2bf(float f) {
  union { float f; unsigned u; } c; c.f = f;
  unsigned u = c.u;
  u += 0x7FFFu + ((u >> 16) & 1u);   // round-to-nearest-even
  return (unsigned short)(u >> 16);
}

// ---------------- f32 -> bf16 conversion (one-time, workspace) ----------------
__global__ void cvt_bf16_kernel(const float* __restrict__ in,
                                unsigned short* __restrict__ out, int n4) {
  int i = blockIdx.x * blockDim.x + threadIdx.x;
  const int stride = gridDim.x * blockDim.x;
  const v4f* in4 = (const v4f*)in;
  v4us* out4 = (v4us*)out;
  for (; i < n4; i += stride) {
    v4f v = in4[i];
    v4us o;
    o.x = f2bf(v.x); o.y = f2bf(v.y); o.z = f2bf(v.z); o.w = f2bf(v.w);
    out4[i] = o;
  }
}

// ---------------- per-token u = scaling[l] * (x[t] @ WA[l]^T), fp32 ----------
__global__ void lora_u_kernel(const float* __restrict__ x,
                              const float* __restrict__ WA,
                              const float* __restrict__ scaling,
                              const int* __restrict__ token_lora,
                              float* __restrict__ U) {
  const int t   = blockIdx.x;
  const int tid = threadIdx.x;
  const int r   = tid >> 4;
  const int j   = tid & 15;
  const int l   = token_lora[t];

  const float* xr = x + (size_t)t * D_IN;
  const float* wr = WA + ((size_t)l * R_RANK + r) * D_IN;

  float acc = 0.f;
#pragma unroll 4
  for (int i = 0; i < D_IN / 64; ++i) {
    const int d = (i * 16 + j) * 4;
    v4f xv = *(const v4f*)(xr + d);
    v4f wv = *(const v4f*)(wr + d);
    acc += xv.x * wv.x + xv.y * wv.y + xv.z * wv.z + xv.w * wv.w;
  }

  __shared__ float red[256];
  red[tid] = acc;
  __syncthreads();
  if (j == 0) {
    float s = 0.f;
#pragma unroll
    for (int jj = 0; jj < 16; ++jj) s += red[(r << 4) + jj];
    U[(size_t)t * R_RANK + r] = scaling[l] * s;
  }
}

// ---------------- main GEMM (bf16 WMMA, f32 accum) + fused bias/LoRA ---------
// 8 wave32s; each wave owns a 64x64 tile (4x4 WMMA subtiles, 128 acc VGPRs).
// Double-buffered LDS, async global->LDS staging when available.
__launch_bounds__(256)
__global__ void gemm_lora_kernel(const unsigned short* __restrict__ xb,
                                 const unsigned short* __restrict__ wb,
                                 const float* __restrict__ b_base,
                                 const float* __restrict__ WB,
                                 const float* __restrict__ U,
                                 const int* __restrict__ token_lora,
                                 float* __restrict__ out) {
  extern __shared__ unsigned short sm[];   // 2 * BUFE halfs

  const int tid  = threadIdx.x;
  const int lane = tid & 31;
  const int wave = tid >> 5;     // 0..7
  const int wm   = wave & 1;     // 0..1  -> 64-row strip
  const int wn   = wave >> 1;    // 0..3  -> 64-col strip
  const int lh   = lane & 15;
  const int sel  = lane >> 4;    // K-half select per WMMA lane layout

  const int rowBase = blockIdx.x * BM;
  const int colBase = blockIdx.y * BN;

  v8f acc[4][4];
#pragma unroll
  for (int i = 0; i < 4; ++i)
#pragma unroll
    for (int j = 0; j < 4; ++j) acc[i][j] = (v8f){0.f,0.f,0.f,0.f,0.f,0.f,0.f,0.f};

  // ---- staging: 384 rows x 64 bf16 (= 3072 16B chunks) by 256 threads ----
  auto stage = [&](int buf, int k0) {
    unsigned short* base = sm + buf * BUFE;
#pragma unroll
    for (int i = 0; i < 4; ++i) {            // x part: 1024 chunks
      const int c = tid + i * 256;
      const int row = c >> 3;
      const int sub = (c & 7) * 8;
      cp16_to_lds(xb + (size_t)(rowBase + row) * D_IN + k0 + sub,
                  base + row * LDSS + sub);
    }
#pragma unroll
    for (int i = 0; i < 8; ++i) {            // w part: 2048 chunks
      const int c = tid + i * 256;
      const int row = c >> 3;
      const int sub = (c & 7) * 8;
      cp16_to_lds(wb + (size_t)(colBase + row) * D_IN + k0 + sub,
                  base + XELEMS + row * LDSS + sub);
    }
  };

  stage(0, 0);   // prologue

  for (int k0 = 0; k0 < D_IN; k0 += BK) {
    const int buf = (k0 >> 6) & 1;
    cp_wait_all();        // my async copies for `buf` done
    __syncthreads();      // everyone's copies done; prior reads of buf^1 done
    if (k0 + BK < D_IN) stage(buf ^ 1, k0 + BK);

    const unsigned short* xs = sm + buf * BUFE;
    const unsigned short* ws = xs + XELEMS;

#pragma unroll
    for (int kk = 0; kk < BK; kk += 32) {
      FragBF a[4], b[4];
#pragma unroll
      for (int mt = 0; mt < 4; ++mt) {
        const unsigned short* ap = xs + (wm * 64 + mt * 16 + lh) * LDSS + kk + sel * 8;
        a[mt].u[0] = *(const v4u*)ap;
        a[mt].u[1] = *(const v4u*)(ap + 16);
      }
#pragma unroll
      for (int nt = 0; nt < 4; ++nt) {
        const unsigned short* bp = ws + (wn * 64 + nt * 16 + lh) * LDSS + kk + sel * 8;
        b[nt].u[0] = *(const v4u*)bp;
        b[nt].u[1] = *(const v4u*)(bp + 16);
      }
#pragma unroll
      for (int mt = 0; mt < 4; ++mt)
#pragma unroll
        for (int nt = 0; nt < 4; ++nt)
          acc[mt][nt] = __builtin_amdgcn_wmma_f32_16x16x32_bf16(
              false, a[mt].v, false, b[nt].v, (short)0, acc[mt][nt], false, false);
    }
  }

  // ---- epilogue: out = acc + bias + u_scaled[row] . WB[l, col, :] ----
#pragma unroll
  for (int mt = 0; mt < 4; ++mt) {
    const int row0 = rowBase + wm * 64 + mt * 16 + sel * 8;
#pragma unroll
    for (int nt = 0; nt < 4; ++nt) {
      const int col = colBase + wn * 64 + nt * 16 + lh;
      const float bb = b_base[col];
      int last_l = -1;
      v4f w0 = {0,0,0,0}, w1 = {0,0,0,0}, w2 = {0,0,0,0}, w3 = {0,0,0,0};
#pragma unroll
      for (int v = 0; v < 8; ++v) {
        const int row = row0 + v;
        const int l = token_lora[row];
        if (l != last_l) {
          const float* wp = WB + ((size_t)l * D_OUT + col) * R_RANK;
          w0 = *(const v4f*)(wp + 0);  w1 = *(const v4f*)(wp + 4);
          w2 = *(const v4f*)(wp + 8);  w3 = *(const v4f*)(wp + 12);
          last_l = l;
        }
        const float* up = U + (size_t)row * R_RANK;
        v4f u0 = *(const v4f*)(up + 0),  u1 = *(const v4f*)(up + 4);
        v4f u2 = *(const v4f*)(up + 8),  u3 = *(const v4f*)(up + 12);
        float s = u0.x*w0.x + u0.y*w0.y + u0.z*w0.z + u0.w*w0.w
                + u1.x*w1.x + u1.y*w1.y + u1.z*w1.z + u1.w*w1.w
                + u2.x*w2.x + u2.y*w2.y + u2.z*w2.z + u2.w*w2.w
                + u3.x*w3.x + u3.y*w3.y + u3.z*w3.z + u3.w*w3.w;
        out[(size_t)row * D_OUT + col] = acc[mt][nt][v] + bb + s;
      }
    }
  }
}

extern "C" void kernel_launch(void* const* d_in, const int* in_sizes, int n_in,
                              void* d_out, int out_size, void* d_ws, size_t ws_size,
                              hipStream_t stream) {
  const float* x          = (const float*)d_in[0];
  const float* W_base     = (const float*)d_in[1];
  const float* b_base     = (const float*)d_in[2];
  const float* WA         = (const float*)d_in[3];
  const float* WB         = (const float*)d_in[4];
  const float* scaling    = (const float*)d_in[5];
  const int*   token_lora = (const int*)d_in[6];
  float* out = (float*)d_out;

  // workspace layout: xb (64MB bf16) | wb (32MB bf16) | U (512KB f32)
  unsigned short* xb = (unsigned short*)d_ws;
  unsigned short* wb = xb + (size_t)T_TOK * D_IN;
  float*          U  = (float*)(wb + (size_t)D_OUT * D_IN);

  cvt_bf16_kernel<<<4096, 256, 0, stream>>>(x,      xb, (T_TOK * D_IN) / 4);
  cvt_bf16_kernel<<<2048, 256, 0, stream>>>(W_base, wb, (D_OUT * D_IN) / 4);
  lora_u_kernel<<<T_TOK, 256, 0, stream>>>(x, WA, scaling, token_lora, U);

  const size_t shmem = (size_t)2 * BUFE * sizeof(unsigned short);  // ~108 KB
  gemm_lora_kernel<<<dim3(T_TOK / BM, D_OUT / BN), 256, shmem, stream>>>(
      xb, wb, b_base, WB, U, token_lora, out);
}
